// GRU_B_7026566496329
// MI455X (gfx1250) — compile-verified, matrix operand order checked by reading, not run on hardware
//
#include <hip/hip_runtime.h>

#define BB 256
#define TT 512
#define FF 64
#define HH 128

typedef __attribute__((ext_vector_type(16))) _Float16 v16h;
typedef __attribute__((ext_vector_type(8)))  _Float16 v8h;
typedef __attribute__((ext_vector_type(8)))  float    v8f;

// Fast transcendentals: single v_exp_f32 (exp2) + single v_rcp_f32 on the
// serial path instead of the IEEE div expansion (div_scale/rcp/fma chain).
__device__ __forceinline__ float sigmoid_f(float x) {
  return __builtin_amdgcn_rcpf(1.0f +
                               __builtin_amdgcn_exp2f(x * -1.442695040888963f));
}
__device__ __forceinline__ float tanh_f(float x) {
  // tanh(x) = 2*sigmoid(2x) - 1
  return 2.0f * __builtin_amdgcn_rcpf(
                    1.0f + __builtin_amdgcn_exp2f(x * -2.885390081777926f)) -
         1.0f;
}

__device__ __forceinline__ v8f wmma16(v16h a, v16h b, v8f c) {
  // D(f32,16x16) = A(f16,16x32) * B(f16,32x16) + C
  return __builtin_amdgcn_wmma_f32_16x16x32_f16(false, a, false, b, (short)0, c,
                                                false, false);
}

// Gather a 32(K) x 16(N) B-fragment from a row-major f32 matrix W[K][ld],
// columns [nBase, nBase+16). VGPR layout per CDNA5 ISA 7.12.2:
//   element (k,n): lane = n%16 + 16*((k%16)/8), vgpr = (k%8)/2 + 4*(k/16), half = k%2
__device__ __forceinline__ v16h load_bfrag(const float* __restrict__ W, int ld,
                                           int nBase, int kb, int lane) {
  const int n  = nBase + (lane & 15);
  const int kh = (lane >> 4) & 1;
  v16h f;
#pragma unroll
  for (int v = 0; v < 8; ++v) {
    int k = kb * 32 + (v & 3) * 2 + 8 * kh + 16 * (v >> 2);
    f[2 * v]     = (_Float16)W[(size_t)k * ld + n];
    f[2 * v + 1] = (_Float16)W[(size_t)(k + 1) * ld + n];
  }
  return f;
}

// Same, but for W^T where W is [N][ld] row-major (lin_w is [F][H]; we need h @ lin_w^T)
__device__ __forceinline__ v16h load_bfrag_T(const float* __restrict__ W, int ld,
                                             int nBase, int kb, int lane) {
  const int n  = nBase + (lane & 15);
  const int kh = (lane >> 4) & 1;
  v16h f;
#pragma unroll
  for (int v = 0; v < 8; ++v) {
    int k = kb * 32 + (v & 3) * 2 + 8 * kh + 16 * (v >> 2);
    f[2 * v]     = (_Float16)W[(size_t)n * ld + k];
    f[2 * v + 1] = (_Float16)W[(size_t)n * ld + k + 1];
  }
  return f;
}

// A-fragment 16(M) x 32(K) f16 from LDS row-major [16][ldh].
// Lane l holds row m=l%16; K halves {0..7,16..23}+8*(l/16)+32*kb
// -> two ds_load_b128 per fragment.
__device__ __forceinline__ v16h load_afrag(const _Float16* A, int ldh, int kb,
                                           int lane) {
  const int m  = lane & 15;
  const int kh = (lane >> 4) & 1;
  const v8h a0 = *(const v8h*)(A + m * ldh + kb * 32 + kh * 8);
  const v8h a1 = *(const v8h*)(A + m * ldh + kb * 32 + 16 + kh * 8);
  v16h r;
#pragma unroll
  for (int i = 0; i < 8; ++i) { r[i] = a0[i]; r[8 + i] = a1[i]; }
  return r;
}

// ---------------------------------------------------------------------------
// Kernel 1: bandwidth-bound precompute of everything off the serial path.
//   pre_g[b,t,:] = (m*x)@w_xg + d@w_dg + m@w_mg + b_g   (t=0: x instead of m*x)
//   cm[b,t,:]    = (1-m)*d                               (t=0: 0)
// ---------------------------------------------------------------------------
__global__ __launch_bounds__(128) void grud_pre(
    const float* __restrict__ X, const float* __restrict__ Mk,
    const float* __restrict__ DL,
    const float* __restrict__ w_xz, const float* __restrict__ w_dz,
    const float* __restrict__ w_mz,
    const float* __restrict__ w_xr, const float* __restrict__ w_dr,
    const float* __restrict__ w_mr,
    const float* __restrict__ w_xh, const float* __restrict__ w_dh,
    const float* __restrict__ w_mh,
    const float* __restrict__ b_z, const float* __restrict__ b_r,
    const float* __restrict__ b_h,
    _Float16* __restrict__ preZ, _Float16* __restrict__ preR,
    _Float16* __restrict__ preH, _Float16* __restrict__ cmh) {
  __shared__ float smx[FF], sdv[FF], smv[FF];
  const size_t bt = blockIdx.x;
  const int t = (int)(bt % TT);
  const int j = threadIdx.x;  // 0..127 = H index
  if (j < FF) {
    size_t o = bt * FF + j;
    float x = X[o], m = Mk[o], dl = DL[o];
    float d = 1.0f - sigmoid_f(dl);
    smv[j] = m;
    sdv[j] = d;
    smx[j] = (t == 0) ? x : m * x;
    cmh[o] = (_Float16)((t == 0) ? 0.0f : (1.0f - m) * d);
  }
  __syncthreads();
  float az = b_z[j], ar = b_r[j], ah = b_h[j];
#pragma unroll 4
  for (int k = 0; k < FF; ++k) {
    float mx = smx[k], d = sdv[k], m = smv[k];
    size_t r = (size_t)k * HH + j;
    az += mx * w_xz[r] + d * w_dz[r] + m * w_mz[r];
    ar += mx * w_xr[r] + d * w_dr[r] + m * w_mr[r];
    ah += mx * w_xh[r] + d * w_dh[r] + m * w_mh[r];
  }
  size_t o = bt * HH + j;
  preZ[o] = (_Float16)az;
  preR[o] = (_Float16)ar;
  preH[o] = (_Float16)ah;
}

// ---------------------------------------------------------------------------
// Kernel 2: serial scan. 16 workgroups x 16 batch rows; 8 waves, one 16-col
// N-tile of H each. Recurrent weights register-resident in WMMA B layout.
// ---------------------------------------------------------------------------
__global__ __launch_bounds__(256, 1) void grud_scan(
    const _Float16* __restrict__ preZ, const _Float16* __restrict__ preR,
    const _Float16* __restrict__ preH, const _Float16* __restrict__ cmh,
    const float* __restrict__ w_xz, const float* __restrict__ w_hz,
    const float* __restrict__ w_xr, const float* __restrict__ w_hr,
    const float* __restrict__ w_xh, const float* __restrict__ w_hh,
    const float* __restrict__ lin_w, const float* __restrict__ lin_b,
    float* __restrict__ out) {
  __shared__ __align__(16) _Float16 hA[16 * HH];   // h_t (f16 copy) as WMMA A
  __shared__ __align__(16) _Float16 rhA[16 * HH];  // r .* h
  __shared__ __align__(16) _Float16 uA[16 * FF];   // (1-m)*d*x_hat

  const int lane  = threadIdx.x & 31;
  const int wave  = threadIdx.x >> 5;           // 0..7
  const int kh    = (lane >> 4) & 1;
  const int nG    = wave * 16 + (lane & 15);    // owned H column
  const int bTile = blockIdx.x;                 // 16 batch rows per WG

  // ---- stationary weight fragments, loaded once, live in VGPRs for all T
  v16h Bhz[4], Bhr[4], Bhh[4], Bxz[2], Bxr[2], Bxh[2], Blin[4];
#pragma unroll
  for (int kb = 0; kb < 4; ++kb) {
    Bhz[kb] = load_bfrag(w_hz, HH, wave * 16, kb, lane);
    Bhr[kb] = load_bfrag(w_hr, HH, wave * 16, kb, lane);
    Bhh[kb] = load_bfrag(w_hh, HH, wave * 16, kb, lane);
  }
#pragma unroll
  for (int kb = 0; kb < 2; ++kb) {
    Bxz[kb] = load_bfrag(w_xz, HH, wave * 16, kb, lane);
    Bxr[kb] = load_bfrag(w_xr, HH, wave * 16, kb, lane);
    Bxh[kb] = load_bfrag(w_xh, HH, wave * 16, kb, lane);
  }
  float linb = 0.0f;
  if (wave < 4) {  // waves 0..3 own the F=64 columns of x_hat
#pragma unroll
    for (int kb = 0; kb < 4; ++kb)
      Blin[kb] = load_bfrag_T(lin_w, HH, wave * 16, kb, lane);
    linb = lin_b[nG];
  }

  v8f hreg = {0.f, 0.f, 0.f, 0.f, 0.f, 0.f, 0.f, 0.f};  // fp32 carried state
  for (int i = threadIdx.x; i < 16 * HH; i += 256) hA[i] = (_Float16)0.0f;
  __syncthreads();

  for (int t = 0; t < TT; ++t) {
    // ---- Phase A: x_hat = sigmoid(h @ lin_w^T + lin_b);  u = cm * x_hat
    if (wave < 4) {
      v8f ax;
#pragma unroll
      for (int v = 0; v < 8; ++v) ax[v] = linb;
#pragma unroll
      for (int kb = 0; kb < 4; ++kb) {
        v16h a = load_afrag(hA, HH, kb, lane);
        ax = wmma16(a, Blin[kb], ax);
      }
#pragma unroll
      for (int v = 0; v < 8; ++v) {
        int m = v + 8 * kh;
        int b = bTile * 16 + m;
        float c = (float)cmh[((size_t)b * TT + t) * FF + nG];
        uA[m * FF + nG] = (_Float16)(c * sigmoid_f(ax[v]));
      }
    } else if (t + 1 < TT) {
      // idle waves: prefetch next step's precomputed gate rows into near cache
      size_t o = ((size_t)(bTile * 16 + 8 * kh) * TT + (t + 1)) * HH + nG;
      __builtin_prefetch(&preZ[o], 0, 3);
      __builtin_prefetch(&preR[o], 0, 3);
      __builtin_prefetch(&preH[o], 0, 3);
    }
    __syncthreads();

    // ---- Phase B: z, r
    v8f az, ar, ah;
#pragma unroll
    for (int v = 0; v < 8; ++v) {
      int m = v + 8 * kh;
      size_t o = ((size_t)(bTile * 16 + m) * TT + t) * HH + nG;
      az[v] = (float)preZ[o];
      ar[v] = (float)preR[o];
      ah[v] = (float)preH[o];
    }
#pragma unroll
    for (int kb = 0; kb < 4; ++kb) {
      v16h a = load_afrag(hA, HH, kb, lane);  // reuse A for both gates
      az = wmma16(a, Bhz[kb], az);
      ar = wmma16(a, Bhr[kb], ar);
    }
#pragma unroll
    for (int kb = 0; kb < 2; ++kb) {
      v16h a = load_afrag(uA, FF, kb, lane);
      az = wmma16(a, Bxz[kb], az);
      ar = wmma16(a, Bxr[kb], ar);
      ah = wmma16(a, Bxh[kb], ah);
    }
    v8f zg, rg;
#pragma unroll
    for (int v = 0; v < 8; ++v) {
      zg[v] = sigmoid_f(az[v]);
      rg[v] = sigmoid_f(ar[v]);
      int m = v + 8 * kh;  // lane owns (m,nG): r*h is lane-local
      rhA[m * HH + nG] = (_Float16)(rg[v] * hreg[v]);
    }
    __syncthreads();

    // ---- h_tilde and state update
#pragma unroll
    for (int kb = 0; kb < 4; ++kb) {
      v16h a = load_afrag(rhA, HH, kb, lane);
      ah = wmma16(a, Bhh[kb], ah);
    }
#pragma unroll
    for (int v = 0; v < 8; ++v) {
      float ht = tanh_f(ah[v]);
      float hn = (1.0f - zg[v]) * hreg[v] + zg[v] * ht;
      hreg[v] = hn;
      int m = v + 8 * kh;
      out[((size_t)(bTile * 16 + m) * TT + t) * HH + nG] = hn;
    }
    __syncthreads();
#pragma unroll
    for (int v = 0; v < 8; ++v) {
      int m = v + 8 * kh;
      hA[m * HH + nG] = (_Float16)hreg[v];
    }
    __syncthreads();
  }
}

// ---------------------------------------------------------------------------
extern "C" void kernel_launch(void* const* d_in, const int* in_sizes, int n_in,
                              void* d_out, int out_size, void* d_ws,
                              size_t ws_size, hipStream_t stream) {
  (void)in_sizes; (void)n_in; (void)out_size; (void)ws_size;
  const float* input = (const float*)d_in[0];
  const float* Mm    = (const float*)d_in[1];
  const float* Delta = (const float*)d_in[2];
  const float* w_xz  = (const float*)d_in[3];
  const float* w_hz  = (const float*)d_in[4];
  const float* w_dz  = (const float*)d_in[5];
  const float* w_mz  = (const float*)d_in[6];
  const float* w_xr  = (const float*)d_in[7];
  const float* w_hr  = (const float*)d_in[8];
  const float* w_dr  = (const float*)d_in[9];
  const float* w_mr  = (const float*)d_in[10];
  const float* w_xh  = (const float*)d_in[11];
  const float* w_hh  = (const float*)d_in[12];
  const float* w_dh  = (const float*)d_in[13];
  const float* w_mh  = (const float*)d_in[14];
  const float* b_z   = (const float*)d_in[15];
  const float* b_r   = (const float*)d_in[16];
  const float* b_h   = (const float*)d_in[17];
  const float* lin_w = (const float*)d_in[18];
  const float* lin_b = (const float*)d_in[19];

  _Float16* ws = (_Float16*)d_ws;
  const size_t BTH = (size_t)BB * TT * HH;
  _Float16* preZ = ws;
  _Float16* preR = ws + BTH;
  _Float16* preH = ws + 2 * BTH;
  _Float16* cmh  = ws + 3 * BTH;

  grud_pre<<<dim3(BB * TT), dim3(HH), 0, stream>>>(
      input, Mm, Delta, w_xz, w_dz, w_mz, w_xr, w_dr, w_mr, w_xh, w_dh, w_mh,
      b_z, b_r, b_h, preZ, preR, preH, cmh);

  grud_scan<<<dim3(BB / 16), dim3(256), 0, stream>>>(
      preZ, preR, preH, cmh, w_xz, w_hz, w_xr, w_hr, w_xh, w_hh, lin_w, lin_b,
      (float*)d_out);
}
